// Attention_50843822850577
// MI455X (gfx1250) — matrix operand order repaired and checked by reading, not compiled, not run on hardware
//
#include <hip/hip_runtime.h>

typedef __attribute__((ext_vector_type(16))) __bf16 v16bf;
typedef __attribute__((ext_vector_type(8)))  float  v8f;

#define Bsz 4
#define Cch 512
#define Nsp 4096

union FragA { uint4 q[2]; v16bf v; };

__device__ __forceinline__ unsigned short f2bf(float f) {
  unsigned int u = __float_as_uint(f);
  u += 0x7FFFu + ((u >> 16) & 1u);   // round to nearest even
  return (unsigned short)(u >> 16);
}

// ---- CDNA5 async global->LDS copy (ASYNCcnt-tracked), GVS addressing -----
// lds_off: 32-bit LDS byte address (VDST), goff: per-lane 32-bit byte offset
// (VADDR), base: uniform 64-bit global base (SADDR).
__device__ __forceinline__ void async_g2l_b128(unsigned lds_off,
                                               unsigned goff,
                                               const void* base) {
  asm volatile("global_load_async_to_lds_b128 %0, %1, %2"
               :
               : "v"(lds_off), "v"(goff), "s"(base)
               : "memory");
}

#if defined(__has_builtin)
#if __has_builtin(__builtin_amdgcn_s_wait_asynccnt)
#define WAIT_ASYNC0() __builtin_amdgcn_s_wait_asynccnt(0)
#endif
#endif
#ifndef WAIT_ASYNC0
#define WAIT_ASYNC0() asm volatile("s_wait_asynccnt 0x0" ::: "memory")
#endif

__device__ __forceinline__ unsigned lds_base_u32(const void* p) {
  return (unsigned)(unsigned long long)p;   // generic addr low 32 bits = LDS offset
}

// ---------------------------------------------------------------- kernel 1
__global__ void __launch_bounds__(256)
an_stats_kernel(const float* __restrict__ content,
                const float* __restrict__ style,
                float* meanC, float* rstdC,
                float* meanS, float* rstdS) {
  const int row = blockIdx.x;          // 0 .. 2*B*C-1
  const int tid = threadIdx.x;
  const int tensor = row >> 11;        // B*C = 2048
  const int rc = row & 2047;
  const float4* p = (const float4*)((tensor ? style : content) + (size_t)rc * Nsp);
  float s = 0.f, s2 = 0.f;
#pragma unroll
  for (int i = 0; i < 4; ++i) {
    float4 v = p[tid + i * 256];
    s  += v.x + v.y + v.z + v.w;
    s2 += v.x * v.x + v.y * v.y + v.z * v.z + v.w * v.w;
  }
  __shared__ float r1[256], r2[256];
  r1[tid] = s; r2[tid] = s2; __syncthreads();
  for (int o = 128; o; o >>= 1) {
    if (tid < o) { r1[tid] += r1[tid + o]; r2[tid] += r2[tid + o]; }
    __syncthreads();
  }
  if (tid == 0) {
    float mean = r1[0] * (1.0f / Nsp);
    float var  = (r2[0] - (float)Nsp * mean * mean) * (1.0f / (Nsp - 1)); // unbiased
    float rs   = rsqrtf(var + 1e-5f);
    if (tensor) { meanS[rc] = mean; rstdS[rc] = rs; }
    else        { meanC[rc] = mean; rstdC[rc] = rs; }
  }
}

// ---------------------------------------------------------------- kernel 2
// Q/K/V projection: out = W * normalize(x) + bias via WMMA bf16.
// Q -> Qt[b][n][c], K -> Kt[b][n][c], V -> Vc[b][c][n]
#define WS 40   // LDS row stride (ushorts): 80B, 16B aligned, conflict-free
__global__ void __launch_bounds__(256)
an_qkv_kernel(const float* __restrict__ content,
              const float* __restrict__ style,
              const float* __restrict__ Wq, const float* __restrict__ bq,
              const float* __restrict__ Wk, const float* __restrict__ bk,
              const float* __restrict__ Wv, const float* __restrict__ bv,
              const float* __restrict__ meanC, const float* __restrict__ rstdC,
              const float* __restrict__ meanS, const float* __restrict__ rstdS,
              unsigned short* Qt, unsigned short* Kt, unsigned short* Vc) {
  __shared__ unsigned short lds_w[64 * WS];
  __shared__ unsigned short lds_x[64 * WS];
  const int tid = threadIdx.x;
  const int n0 = blockIdx.x * 64, c0 = blockIdx.y * 64;
  const int which = blockIdx.z % 3, b = blockIdx.z / 3;
  const float* W    = which == 0 ? Wq : (which == 1 ? Wk : Wv);
  const float* bias = which == 0 ? bq : (which == 1 ? bk : bv);
  const float* x    = which == 0 ? content : style;
  const float* meanA = which == 0 ? meanC : meanS;
  const float* rstdA = which == 0 ? rstdC : rstdS;
  const int wave = tid >> 5, lane = tid & 31, half = lane >> 4, lq = lane & 15;
  const int waveM = wave >> 1, waveN = wave & 1;

  v8f acc0, acc1;
#pragma unroll
  for (int r = 0; r < 8; ++r) {               // bias pre-loaded into C
    float bvv = bias[c0 + waveM * 16 + half * 8 + r];
    acc0[r] = bvv; acc1[r] = bvv;
  }

  for (int ks = 0; ks < 16; ++ks) {
    const int k0 = ks * 32;
    __syncthreads();
    // stage W tile [64 cout x 32 k] as bf16
#pragma unroll
    for (int i = 0; i < 2; ++i) {
      int row = i * 32 + (tid >> 3), grp = tid & 7;
      float4 w4 = *(const float4*)(W + (size_t)(c0 + row) * Cch + k0 + grp * 4);
      unsigned short* dst = &lds_w[row * WS + grp * 4];
      dst[0] = f2bf(w4.x); dst[1] = f2bf(w4.y); dst[2] = f2bf(w4.z); dst[3] = f2bf(w4.w);
    }
    // stage X tile transposed [64 n x 32 k], normalized (Q/K) or raw (V)
#pragma unroll
    for (int i = 0; i < 2; ++i) {
      int row = i * 16 + (tid >> 4), g = tid & 15;
      float mu = 0.f, rs = 1.f;
      if (which < 2) { mu = meanA[b * Cch + k0 + row]; rs = rstdA[b * Cch + k0 + row]; }
      float4 xv = *(const float4*)(x + ((size_t)(b * Cch + k0 + row)) * Nsp + n0 + g * 4);
      lds_x[(g * 4 + 0) * WS + row] = f2bf((xv.x - mu) * rs);
      lds_x[(g * 4 + 1) * WS + row] = f2bf((xv.y - mu) * rs);
      lds_x[(g * 4 + 2) * WS + row] = f2bf((xv.z - mu) * rs);
      lds_x[(g * 4 + 3) * WS + row] = f2bf((xv.w - mu) * rs);
    }
    __syncthreads();
    FragA A, B0, B1;
    {
      int row = waveM * 16 + lq;
      A.q[0] = *(const uint4*)&lds_w[row * WS + half * 16];
      A.q[1] = *(const uint4*)&lds_w[row * WS + half * 16 + 8];
    }
    {
      int row = waveN * 32 + lq;
      B0.q[0] = *(const uint4*)&lds_x[row * WS + half * 16];
      B0.q[1] = *(const uint4*)&lds_x[row * WS + half * 16 + 8];
      row += 16;
      B1.q[0] = *(const uint4*)&lds_x[row * WS + half * 16];
      B1.q[1] = *(const uint4*)&lds_x[row * WS + half * 16 + 8];
    }
    acc0 = __builtin_amdgcn_wmma_f32_16x16x32_bf16(false, A.v, false, B0.v, (short)0, acc0, false, false);
    acc1 = __builtin_amdgcn_wmma_f32_16x16x32_bf16(false, A.v, false, B1.v, (short)0, acc1, false, false);
  }

  const int cbase = c0 + waveM * 16 + half * 8;
  if (which < 2) {        // Q,K -> [n][c] transposed store, 16B vector per lane
    unsigned short* T = which == 0 ? Qt : Kt;
#pragma unroll
    for (int nt = 0; nt < 2; ++nt) {
      v8f a = nt ? acc1 : acc0;
      int n = n0 + waveN * 32 + nt * 16 + lq;
      uint4 pk;
      pk.x = (unsigned)f2bf(a[0]) | ((unsigned)f2bf(a[1]) << 16);
      pk.y = (unsigned)f2bf(a[2]) | ((unsigned)f2bf(a[3]) << 16);
      pk.z = (unsigned)f2bf(a[4]) | ((unsigned)f2bf(a[5]) << 16);
      pk.w = (unsigned)f2bf(a[6]) | ((unsigned)f2bf(a[7]) << 16);
      *(uint4*)(T + ((size_t)(b * Nsp + n)) * Cch + cbase) = pk;
    }
  } else {                // V -> [c][n] channel-major store
#pragma unroll
    for (int nt = 0; nt < 2; ++nt) {
      v8f a = nt ? acc1 : acc0;
      int n = n0 + waveN * 32 + nt * 16 + lq;
#pragma unroll
      for (int r = 0; r < 8; ++r)
        Vc[((size_t)(b * Cch + cbase + r)) * Nsp + n] = f2bf(a[r]);
    }
  }
}

// ---------------------------------------------------------------- kernel 3
// Pass 1: per-row softmax stats (max, 1/sum) via flash-style online reduction.
#define QS 520  // Q LDS row stride (ushorts); row stride = 1040B (16B aligned)
__global__ void __launch_bounds__(256)
an_softmax_stats_kernel(const unsigned short* __restrict__ Qt,
                        const unsigned short* __restrict__ Kt,
                        float* rowMax, float* rowInvL) {
  __shared__ unsigned short sQ[16 * QS];
  __shared__ float sMax[8 * 16], sSum[8 * 16];
  const int tid = threadIdx.x;
  const int b = blockIdx.y, n0 = blockIdx.x * 16;
  const int wave = tid >> 5, lane = tid & 31, half = lane >> 4, lq = lane & 15;
  // stage 16x512 Q tile (one contiguous 16KB block) via async DMA to LDS
  {
    const void* qbase = (const void*)(Qt + ((size_t)(b * Nsp + n0)) * Cch);
    const unsigned sqb = lds_base_u32(sQ);
#pragma unroll
    for (int i = 0; i < 4; ++i) {
      unsigned id = tid + i * 256;                       // 16B chunk id
      unsigned goff = id * 16;                           // contiguous global
      unsigned loff = sqb + (id >> 6) * (QS * 2) + (id & 63) * 16; // padded rows
      async_g2l_b128(loff, goff, qbase);
    }
    WAIT_ASYNC0();
  }
  __syncthreads();
  float mu[8], sg[8];
#pragma unroll
  for (int r = 0; r < 8; ++r) { mu[r] = -1e30f; sg[r] = 0.f; }

  for (int m0 = wave * 16; m0 < Nsp; m0 += 128) {
    const unsigned short* kt = Kt + ((size_t)(b * Nsp + m0 + lq)) * Cch + half * 16;
    __builtin_prefetch(kt + 128 * Cch, 0, 1);  // next key block -> global_prefetch
    v8f s;
#pragma unroll
    for (int r = 0; r < 8; ++r) s[r] = 0.f;
#pragma unroll
    for (int ks = 0; ks < 16; ++ks) {
      FragA A, Bf;
      A.q[0]  = *(const uint4*)&sQ[lq * QS + ks * 32 + half * 16];
      A.q[1]  = *(const uint4*)&sQ[lq * QS + ks * 32 + half * 16 + 8];
      Bf.q[0] = *(const uint4*)(kt + ks * 32);
      Bf.q[1] = *(const uint4*)(kt + ks * 32 + 8);
      s = __builtin_amdgcn_wmma_f32_16x16x32_bf16(false, A.v, false, Bf.v, (short)0, s, false, false);
    }
#pragma unroll
    for (int r = 0; r < 8; ++r) {   // row reduce across the 16-lane group
      float tm = s[r];
      tm = fmaxf(tm, __shfl_xor(tm, 1, 32));
      tm = fmaxf(tm, __shfl_xor(tm, 2, 32));
      tm = fmaxf(tm, __shfl_xor(tm, 4, 32));
      tm = fmaxf(tm, __shfl_xor(tm, 8, 32));
      float te = __expf(s[r] - tm);
      te += __shfl_xor(te, 1, 32);
      te += __shfl_xor(te, 2, 32);
      te += __shfl_xor(te, 4, 32);
      te += __shfl_xor(te, 8, 32);
      float nm = fmaxf(mu[r], tm);
      sg[r] = sg[r] * __expf(mu[r] - nm) + te * __expf(tm - nm);
      mu[r] = nm;
    }
  }
  if (lq == 0) {   // lanes 0 and 16 publish per-row stats
#pragma unroll
    for (int r = 0; r < 8; ++r) {
      sMax[wave * 16 + half * 8 + r] = mu[r];
      sSum[wave * 16 + half * 8 + r] = sg[r];
    }
  }
  __syncthreads();
  if (tid < 16) {  // merge the 8 wave-local online states
    float M = -1e30f;
#pragma unroll
    for (int w = 0; w < 8; ++w) M = fmaxf(M, sMax[w * 16 + tid]);
    float L = 0.f;
#pragma unroll
    for (int w = 0; w < 8; ++w) L += sSum[w * 16 + tid] * __expf(sMax[w * 16 + tid] - M);
    rowMax [b * Nsp + n0 + tid] = M;
    rowInvL[b * Nsp + n0 + tid] = 1.0f / L;
  }
}

// ---------------------------------------------------------------- kernel 4
// Pass 2: Out = (1/L) * exp(Q^T K - M) @ V^T, waves own 64-channel chunks.
#define PS 136  // P LDS row stride (ushorts)
__global__ void __launch_bounds__(256)
an_attn_out_kernel(const unsigned short* __restrict__ Qt,
                   const unsigned short* __restrict__ Kt,
                   const unsigned short* __restrict__ Vc,
                   const float* __restrict__ rowMax,
                   const float* __restrict__ rowInvL,
                   float* __restrict__ out) {
  __shared__ unsigned short sQ[16 * QS];
  __shared__ unsigned short sP[16 * PS];
  const int tid = threadIdx.x;
  const int b = blockIdx.y, n0 = blockIdx.x * 16;
  const int wave = tid >> 5, lane = tid & 31, half = lane >> 4, lq = lane & 15;
  // async DMA the contiguous 16KB Q tile into padded LDS rows
  {
    const void* qbase = (const void*)(Qt + ((size_t)(b * Nsp + n0)) * Cch);
    const unsigned sqb = lds_base_u32(sQ);
#pragma unroll
    for (int i = 0; i < 4; ++i) {
      unsigned id = tid + i * 256;
      unsigned goff = id * 16;
      unsigned loff = sqb + (id >> 6) * (QS * 2) + (id & 63) * 16;
      async_g2l_b128(loff, goff, qbase);
    }
    WAIT_ASYNC0();
  }
  float Mr[8], iL[8];
#pragma unroll
  for (int r = 0; r < 8; ++r) {
    Mr[r] = rowMax [b * Nsp + n0 + half * 8 + r];
    iL[r] = rowInvL[b * Nsp + n0 + half * 8 + r];
  }
  v8f acc[4];
#pragma unroll
  for (int ct = 0; ct < 4; ++ct)
#pragma unroll
    for (int r = 0; r < 8; ++r) acc[ct][r] = 0.f;
  const int c0 = wave * 64;
  __syncthreads();

  for (int mb = 0; mb < Nsp; mb += 128) {
    // Phase A: cooperative S -> P (each wave computes 16 key columns)
    const int m0 = mb + wave * 16;
    const unsigned short* kt = Kt + ((size_t)(b * Nsp + m0 + lq)) * Cch + half * 16;
    __builtin_prefetch(kt + 128 * Cch, 0, 1);
    v8f s;
#pragma unroll
    for (int r = 0; r < 8; ++r) s[r] = 0.f;
#pragma unroll
    for (int ks = 0; ks < 16; ++ks) {
      FragA A, Bf;
      A.q[0]  = *(const uint4*)&sQ[lq * QS + ks * 32 + half * 16];
      A.q[1]  = *(const uint4*)&sQ[lq * QS + ks * 32 + half * 16 + 8];
      Bf.q[0] = *(const uint4*)(kt + ks * 32);
      Bf.q[1] = *(const uint4*)(kt + ks * 32 + 8);
      s = __builtin_amdgcn_wmma_f32_16x16x32_bf16(false, A.v, false, Bf.v, (short)0, s, false, false);
    }
#pragma unroll
    for (int r = 0; r < 8; ++r)
      sP[(half * 8 + r) * PS + wave * 16 + lq] = f2bf(__expf(s[r] - Mr[r]));
    __syncthreads();
    // Phase B: acc += P(16x128) @ V^T(128 x 64-chunk)
#pragma unroll
    for (int ks = 0; ks < 4; ++ks) {
      FragA A;
      A.q[0] = *(const uint4*)&sP[lq * PS + ks * 32 + half * 16];
      A.q[1] = *(const uint4*)&sP[lq * PS + ks * 32 + half * 16 + 8];
#pragma unroll
      for (int ct = 0; ct < 4; ++ct) {
        const unsigned short* vp =
            Vc + ((size_t)(b * Cch + c0 + ct * 16 + lq)) * Nsp + mb + ks * 32 + half * 16;
        FragA Bf;
        Bf.q[0] = *(const uint4*)vp;
        Bf.q[1] = *(const uint4*)(vp + 8);
        acc[ct] = __builtin_amdgcn_wmma_f32_16x16x32_bf16(false, A.v, false, Bf.v, (short)0, acc[ct], false, false);
      }
    }
    __syncthreads();   // protect sP before next key block
  }
#pragma unroll
  for (int ct = 0; ct < 4; ++ct) {
    float* dst = out + ((size_t)(b * Cch + c0 + ct * 16 + lq)) * Nsp + n0 + half * 8;
    float4 f0 = { acc[ct][0] * iL[0], acc[ct][1] * iL[1], acc[ct][2] * iL[2], acc[ct][3] * iL[3] };
    float4 f1 = { acc[ct][4] * iL[4], acc[ct][5] * iL[5], acc[ct][6] * iL[6], acc[ct][7] * iL[7] };
    *(float4*)dst = f0;
    *(float4*)(dst + 4) = f1;
  }
}

// ---------------------------------------------------------------- launch
extern "C" void kernel_launch(void* const* d_in, const int* in_sizes, int n_in,
                              void* d_out, int out_size, void* d_ws, size_t ws_size,
                              hipStream_t stream) {
  (void)in_sizes; (void)n_in; (void)out_size; (void)ws_size;
  const float* content = (const float*)d_in[0];
  const float* style   = (const float*)d_in[1];
  const float* Wq = (const float*)d_in[2];
  const float* bq = (const float*)d_in[3];
  const float* Wk = (const float*)d_in[4];
  const float* bk = (const float*)d_in[5];
  const float* Wv = (const float*)d_in[6];
  const float* bv = (const float*)d_in[7];

  char* ws = (char*)d_ws;
  float* meanC = (float*)ws;            // [B*C]
  float* rstdC = meanC + Bsz * Cch;
  float* meanS = rstdC + Bsz * Cch;
  float* rstdS = meanS + Bsz * Cch;
  size_t off = 32768;
  unsigned short* Qt = (unsigned short*)(ws + off); off += (size_t)Bsz * Nsp * Cch * 2;
  unsigned short* Kt = (unsigned short*)(ws + off); off += (size_t)Bsz * Nsp * Cch * 2;
  unsigned short* Vc = (unsigned short*)(ws + off); off += (size_t)Bsz * Nsp * Cch * 2;
  float* rowMax  = (float*)(ws + off);  off += (size_t)Bsz * Nsp * 4;
  float* rowInvL = (float*)(ws + off);

  an_stats_kernel<<<dim3(2 * Bsz * Cch), 256, 0, stream>>>(
      content, style, meanC, rstdC, meanS, rstdS);
  an_qkv_kernel<<<dim3(Nsp / 64, Cch / 64, 3 * Bsz), 256, 0, stream>>>(
      content, style, Wq, bq, Wk, bk, Wv, bv,
      meanC, rstdC, meanS, rstdS, Qt, Kt, Vc);
  an_softmax_stats_kernel<<<dim3(Nsp / 16, Bsz), 256, 0, stream>>>(
      Qt, Kt, rowMax, rowInvL);
  an_attn_out_kernel<<<dim3(Nsp / 16, Bsz), 256, 0, stream>>>(
      Qt, Kt, Vc, rowMax, rowInvL, (float*)d_out);
}
